// RTERModel_50577534878402
// MI455X (gfx1250) — compile-verified, matrix operand order checked by reading, not run on hardware
//
#include <hip/hip_runtime.h>
#include <stdint.h>

// ---------------------------------------------------------------------------
// Types for CDNA5 WMMA (bf16 in, f32 accumulate)
// ---------------------------------------------------------------------------
typedef __bf16 bf16;
typedef __attribute__((ext_vector_type(16))) bf16  v16bf;
typedef __attribute__((ext_vector_type(8)))  float v8f;

#define HDIM  300
#define HPAD  304      // H padded to 16
#define KPAD  320      // H padded to 32 (WMMA K)
#define K2PAD 640      // 2H padded to 32
#define G3    912      // 3 * HPAD (r,z,n gate-packed columns)
#define G2    608      // 2 * HPAD (attn r,n)
#define NB    64       // dialogue length N
#define TT    128      // tokens T
#define WN    40       // context window
#define NC1   63       // N-1
#define O2STR 608      // layer-1 output stride (2H padded to 8)
#define NT_H  19       // HPAD/16 column tiles

#define CEILDIV(a,b) (((a)+(b)-1)/(b))

// ---------------------------------------------------------------------------
// Fragment load: 16-bit A/B fragment for V_WMMA_*_16X16X32.
// Lane L: row/col = L&15 ; K chunk: lanes<16 -> K[0..7]+[16..23],
// lanes>=16 -> K[8..15]+[24..31]  (two 16B contiguous loads).
// ---------------------------------------------------------------------------
__device__ inline v16bf load_frag(const bf16* base, int stride, int lane) {
  int r  = lane & 15;
  int ko = (lane >> 4) << 3;           // 0 or 8
  const bf16* p = base + (size_t)r * stride + ko;
  union { v16bf v; uint4 q[2]; } f;
  f.q[0] = *(const uint4*)(p);         // K[ko .. ko+7]
  f.q[1] = *(const uint4*)(p + 16);    // K[ko+16 .. ko+23]
  return f.v;
}

__device__ inline v8f wmma_bf16(v16bf a, v16bf b, v8f c) {
  return __builtin_amdgcn_wmma_f32_16x16x32_bf16(false, a, false, b,
                                                 (short)0, c, false, false);
}

__device__ inline float sigmoidf(float x) { return 1.f / (1.f + __expf(-x)); }

// ---------------------------------------------------------------------------
// Gate-packed WMMA GEMM:
//   C[M, NG*HPAD] = A[M,K] * B[NG*HPAD, K]^T + bias[NG*HPAD]
// One wave computes NG gate tiles at (mt, nt), reusing the A fragment NG
// times (3 independent WMMA accumulation chains for the GRU case).
// grid = (M/16, NT_H), block = 32.
// ---------------------------------------------------------------------------
template <int NG>
__global__ __launch_bounds__(32)
void wmma_gemm_bias(const bf16* __restrict__ A, int lda,
                    const bf16* __restrict__ B, int ldb,
                    const float* __restrict__ bias,
                    float* __restrict__ C, int ldc, int kchunks) {
  int mt = blockIdx.x, nt = blockIdx.y;
  int lane = threadIdx.x;
  v8f acc[NG];
#pragma unroll
  for (int g = 0; g < NG; ++g) acc[g] = (v8f){};
  const bf16* arow = A + (size_t)mt * 16 * lda;
  for (int kc = 0; kc < kchunks; ++kc) {
    v16bf a = load_frag(arow + kc * 32, lda, lane);
#pragma unroll
    for (int g = 0; g < NG; ++g) {
      v16bf b = load_frag(B + ((size_t)(g * HPAD + nt * 16)) * ldb + kc * 32, ldb, lane);
      acc[g] = wmma_bf16(a, b, acc[g]);
    }
  }
  int n  = nt * 16 + (lane & 15);
  int m0 = mt * 16 + ((lane >> 4) << 3);
#pragma unroll
  for (int g = 0; g < NG; ++g) {
    float bv = bias[g * HPAD + n];
#pragma unroll
    for (int i = 0; i < 8; ++i)
      C[(size_t)(m0 + i) * ldc + g * HPAD + n] = acc[g][i] + bv;
  }
}

// ---------------------------------------------------------------------------
// Persistent GRU scan. One block per direction (blockIdx.x), 1024 thr = 32 waves.
// h kept in LDS: f32 master + bf16 WMMA copy. Per step computes
// gh = h @ Whh^T (WMMA, gates packed 3x304) then gate math in f32.
// gi (precomputed x@Wih^T + bih) is read from global: [T][NB][G3].
// Next step's gi slab is prefetched (global_prefetch_b8) during compute.
// ---------------------------------------------------------------------------
struct ScanArgs {
  const float* gi[2];
  const bf16*  whh[2];
  const float* bhh[2];
  const int*   slens;
  float* of32[2]; int of32_stride; int of32_coff[2];
  bf16*  obf[2];  int obf_stride;  int obf_coff[2];
  int T;
  int use_mask;
};

__global__ __launch_bounds__(1024)
void gru_scan(ScanArgs args) {
  __shared__ float s_hf[NB][HPAD];
  __shared__ bf16  s_hb[NB][KPAD];
  __shared__ int   s_len[NB];
  const int dir = blockIdx.x;
  const int tid = threadIdx.x;
  for (int i = tid; i < NB * HPAD; i += blockDim.x) ((float*)s_hf)[i] = 0.f;
  for (int i = tid; i < NB * KPAD; i += blockDim.x) ((bf16*)s_hb)[i] = (bf16)0.f;
  if (tid < NB) s_len[tid] = args.use_mask ? args.slens[tid] : 0x7fffffff;
  __syncthreads();

  const int wid = tid >> 5, lane = tid & 31;
  const float* gi  = args.gi[dir];
  const bf16*  whh = args.whh[dir];
  const float* bhh = args.bhh[dir];
  const int NTILES = 4 * NT_H;  // 76
  float hnew_buf[3][8];

  for (int ti = 0; ti < args.T; ++ti) {
    const int t = dir ? (args.T - 1 - ti) : ti;
    // Prefetch next timestep's gate slab into cache while we compute.
    {
      const int tn = dir ? (args.T - 2 - ti) : (ti + 1);
      if (tn >= 0 && tn < args.T) {
        const float* nxt = gi + (size_t)tn * NB * G3;
        for (int off = tid * 16; off < NB * G3; off += (int)blockDim.x * 16)
          __builtin_prefetch(nxt + off, 0, 0);
      }
    }
    int slot = 0;
    for (int tile = wid; tile < NTILES; tile += 32, ++slot) {
      int mt = tile / NT_H, nt = tile % NT_H;
      v8f ar = {}, az = {}, an = {};
      for (int kc = 0; kc < 10; ++kc) {
        v16bf a  = load_frag(&s_hb[mt * 16][kc * 32], KPAD, lane);
        v16bf br = load_frag(whh + ((size_t)(0 * HPAD + nt * 16)) * KPAD + kc * 32, KPAD, lane);
        v16bf bz = load_frag(whh + ((size_t)(1 * HPAD + nt * 16)) * KPAD + kc * 32, KPAD, lane);
        v16bf bn = load_frag(whh + ((size_t)(2 * HPAD + nt * 16)) * KPAD + kc * 32, KPAD, lane);
        ar = wmma_bf16(a, br, ar);
        az = wmma_bf16(a, bz, az);
        an = wmma_bf16(a, bn, an);
      }
      int n  = nt * 16 + (lane & 15);
      int m0 = mt * 16 + ((lane >> 4) << 3);
      float b_r = bhh[n], b_z = bhh[HPAD + n], b_n = bhh[2 * HPAD + n];
      const float* gslab = gi + (size_t)t * NB * G3;
      for (int i = 0; i < 8; ++i) {
        int m = m0 + i;
        const float* g = gslab + (size_t)m * G3;
        float r  = sigmoidf(g[n] + ar[i] + b_r);
        float z  = sigmoidf(g[HPAD + n] + az[i] + b_z);
        float nn = tanhf(g[2 * HPAD + n] + an[i] + b_n);
        float hold = s_hf[m][n];
        float hnew = (1.f - z) * nn + z * hold;
        float ov   = hnew;
        if (args.use_mask && t >= s_len[m]) { hnew = hold; ov = 0.f; }
        hnew_buf[slot][i] = hnew;
        if (n < HDIM) {
          size_t orow = (size_t)t * NB + m;
          if (args.of32[dir])
            args.of32[dir][orow * args.of32_stride + args.of32_coff[dir] + n] = ov;
          if (args.obf[dir])
            args.obf[dir][orow * args.obf_stride + args.obf_coff[dir] + n] = (bf16)ov;
        }
      }
    }
    __syncthreads();
    slot = 0;
    for (int tile = wid; tile < NTILES; tile += 32, ++slot) {
      int mt = tile / NT_H, nt = tile % NT_H;
      int n  = nt * 16 + (lane & 15);
      int m0 = mt * 16 + ((lane >> 4) << 3);
      if (n < HDIM)
        for (int i = 0; i < 8; ++i) {
          s_hf[m0 + i][n] = hnew_buf[slot][i];
          s_hb[m0 + i][n] = (bf16)hnew_buf[slot][i];
        }
    }
    __syncthreads();
  }
}

// ---------------------------------------------------------------------------
// Attention-GRU scan (AGRU). ci = c@{Wr,Wn}^T + {br,bw} precomputed [WN][NB][G2].
// Per step: gh = h @ {Ur,Un}^T (+{bur,bu}); r=sig(ci_r+gh_r);
// ht=tanh(ci_n + r*gh_n); h = g*ht + (1-g)*h. Writes final h.
// ---------------------------------------------------------------------------
struct AttnArgs {
  const float* ci[2];
  const bf16*  u[2];
  const float* bu[2];
  const float* gates;    // [WN][NB]
  float* hout[2];        // [NB][HPAD]
  int T;
};

__global__ __launch_bounds__(1024)
void attn_scan(AttnArgs args) {
  __shared__ float s_hf[NB][HPAD];
  __shared__ bf16  s_hb[NB][KPAD];
  const int dir = blockIdx.x;
  const int tid = threadIdx.x;
  for (int i = tid; i < NB * HPAD; i += blockDim.x) ((float*)s_hf)[i] = 0.f;
  for (int i = tid; i < NB * KPAD; i += blockDim.x) ((bf16*)s_hb)[i] = (bf16)0.f;
  __syncthreads();

  const int wid = tid >> 5, lane = tid & 31;
  const float* ci = args.ci[dir];
  const bf16*  u  = args.u[dir];
  const float* bu = args.bu[dir];
  float hnew_buf[3][8];

  for (int ti = 0; ti < args.T; ++ti) {
    const int t = dir ? (args.T - 1 - ti) : ti;
    {
      const int tn = dir ? (args.T - 2 - ti) : (ti + 1);
      if (tn >= 0 && tn < args.T) {
        const float* nxt = ci + (size_t)tn * NB * G2;
        for (int off = tid * 16; off < NB * G2; off += (int)blockDim.x * 16)
          __builtin_prefetch(nxt + off, 0, 0);
      }
    }
    int slot = 0;
    for (int tile = wid; tile < 4 * NT_H; tile += 32, ++slot) {
      int mt = tile / NT_H, nt = tile % NT_H;
      v8f arr = {}, ann = {};
      for (int kc = 0; kc < 10; ++kc) {
        v16bf a  = load_frag(&s_hb[mt * 16][kc * 32], KPAD, lane);
        v16bf br = load_frag(u + ((size_t)(nt * 16)) * KPAD + kc * 32, KPAD, lane);
        v16bf bn = load_frag(u + ((size_t)(HPAD + nt * 16)) * KPAD + kc * 32, KPAD, lane);
        arr = wmma_bf16(a, br, arr);
        ann = wmma_bf16(a, bn, ann);
      }
      int n  = nt * 16 + (lane & 15);
      int m0 = mt * 16 + ((lane >> 4) << 3);
      float b_r = bu[n], b_n = bu[HPAD + n];
      const float* cslab = ci + (size_t)t * NB * G2;
      for (int i = 0; i < 8; ++i) {
        int m = m0 + i;
        const float* c = cslab + (size_t)m * G2;
        float r  = sigmoidf(c[n] + arr[i] + b_r);
        float ht = tanhf(c[HPAD + n] + r * (ann[i] + b_n));
        float g  = args.gates[t * NB + m];
        hnew_buf[slot][i] = g * ht + (1.f - g) * s_hf[m][n];
      }
    }
    __syncthreads();
    slot = 0;
    for (int tile = wid; tile < 4 * NT_H; tile += 32, ++slot) {
      int mt = tile / NT_H, nt = tile % NT_H;
      int n  = nt * 16 + (lane & 15);
      int m0 = mt * 16 + ((lane >> 4) << 3);
      if (n < HDIM)
        for (int i = 0; i < 8; ++i) {
          s_hf[m0 + i][n] = hnew_buf[slot][i];
          s_hb[m0 + i][n] = (bf16)hnew_buf[slot][i];
        }
    }
    __syncthreads();
  }
  for (int i = tid; i < NB * HPAD; i += blockDim.x)
    args.hout[dir][i] = ((float*)s_hf)[i];
}

// ---------------------------------------------------------------------------
// Small glue kernels
// ---------------------------------------------------------------------------
__global__ void k_fill_zero_bf16(bf16* p, int n) {
  int i = blockIdx.x * blockDim.x + threadIdx.x;
  if (i < n) p[i] = (bf16)0.f;
}

// Pack fp32 weights [ngates*rows][cols] -> bf16 [ngates*rpad][kpad], zero pads.
__global__ void k_pack_w(const float* __restrict__ src, bf16* __restrict__ dst,
                         int rows, int rpad, int cols, int kpad, int ngates) {
  int idx = blockIdx.x * blockDim.x + threadIdx.x;
  int total = ngates * rpad * kpad;
  if (idx >= total) return;
  int c = idx % kpad;
  int r = idx / kpad;
  int g = r / rpad;
  int i = r % rpad;
  float v = 0.f;
  if (i < rows && c < cols) v = src[((size_t)(g * rows + i)) * cols + c];
  dst[idx] = (bf16)v;
}

__global__ void k_pack_b(const float* __restrict__ src, float* __restrict__ dst,
                         int rows, int rpad, int ngates) {
  int idx = blockIdx.x * blockDim.x + threadIdx.x;
  if (idx >= ngates * rpad) return;
  int g = idx / rpad, i = idx % rpad;
  dst[idx] = (i < rows) ? src[g * rows + i] : 0.f;
}

// Embedding gather -> X0 bf16 [TT*NB][KPAD]; row = t*NB+n uses ids[n][t].
__global__ void k_embed(const int* __restrict__ ids, const float* __restrict__ table,
                        bf16* __restrict__ X) {
  int idx = blockIdx.x * blockDim.x + threadIdx.x;
  if (idx >= TT * NB * KPAD) return;
  int c = idx % KPAD;
  int row = idx / KPAD;
  int n = row % NB, t = row / NB;
  float v = 0.f;
  if (c < HDIM) {
    int id = ids[n * TT + t];
    v = table[(size_t)id * HDIM + c];
  }
  X[idx] = (bf16)v;
}

// Max over time of layer-1 output (masked steps already 0) -> bf16 [NB][K2PAD]
__global__ void k_maxpool(const float* __restrict__ o2, bf16* __restrict__ uttmax) {
  int idx = blockIdx.x * blockDim.x + threadIdx.x;
  if (idx >= NB * K2PAD) return;
  int c = idx % K2PAD, b = idx / K2PAD;
  float v = 0.f;
  if (c < 2 * HDIM) {
    v = -1e30f;
    for (int t = 0; t < TT; ++t)
      v = fmaxf(v, o2[((size_t)t * NB + b) * O2STR + c]);
  }
  uttmax[idx] = (bf16)v;
}

__global__ void k_tanh_utt(const float* __restrict__ pre, float* __restrict__ utt) {
  int idx = blockIdx.x * blockDim.x + threadIdx.x;
  if (idx >= NB * HPAD) return;
  utt[idx] = tanhf(pre[idx]);     // pads: tanh(0)=0
}

// batches[j][i][c] = (i<63 && i+1-WN+j>=0) ? utt[i+1-WN+j][c] : 0   (bf16, padded)
__global__ void k_batches(const float* __restrict__ utt, bf16* __restrict__ batches) {
  int idx = blockIdx.x * blockDim.x + threadIdx.x;
  if (idx >= WN * NB * KPAD) return;
  int c = idx % KPAD;
  int rem = idx / KPAD;
  int i = rem % NB, j = rem / NB;
  float v = 0.f;
  if (i < NC1 && c < HDIM) {
    int src = i + 1 - WN + j;
    if (src >= 0) v = utt[(size_t)src * HPAD + c];
  }
  batches[idx] = (bf16)v;
}

// mem = batches + f + b  (f32 [WN][NB][HPAD] and bf16 [WN][NB][KPAD], zero pads)
__global__ void k_msum(const float* __restrict__ utt, const float* __restrict__ fctx,
                       const float* __restrict__ bctx,
                       float* __restrict__ msf, bf16* __restrict__ msb) {
  int idx = blockIdx.x * blockDim.x + threadIdx.x;
  if (idx >= WN * NB * KPAD) return;
  int c = idx % KPAD;
  int rem = idx / KPAD;
  int i = rem % NB, j = rem / NB;
  float v = 0.f;
  if (i < NC1 && c < HDIM) {
    int src = i + 1 - WN + j;
    float bv = (src >= 0) ? utt[(size_t)src * HPAD + c] : 0.f;
    size_t o = ((size_t)j * NB + i) * HPAD + c;
    v = bv + fctx[o] + bctx[o];
  }
  msb[idx] = (bf16)v;
  if (c < HPAD) msf[((size_t)j * NB + i) * HPAD + c] = v;
}

__global__ void k_qinit(const float* __restrict__ utt, float* __restrict__ q) {
  int idx = blockIdx.x * blockDim.x + threadIdx.x;
  if (idx >= NB * HPAD) return;
  int b = idx / HPAD;
  int c = idx % HPAD;
  q[idx] = (b < NC1) ? utt[(size_t)(b + 1) * HPAD + c] : 0.f;
}

// scores + masked softmax -> gates [WN][NB]
__global__ __launch_bounds__(64)
void k_scores(const float* __restrict__ q, const float* __restrict__ msf,
              float* __restrict__ gates) {
  int b = blockIdx.x;     // 0..63
  int w = threadIdx.x;    // 0..63
  __shared__ float sc[WN];
  float s = -1e10f;
  if (b < NC1 && w < WN && w >= (WN - 1 - b)) {
    float acc = 0.f;
    for (int c = 0; c < HDIM; ++c)
      acc += q[(size_t)b * HPAD + c] * msf[((size_t)w * NB + b) * HPAD + c];
    s = acc;
  }
  if (w < WN) sc[w] = s;
  __syncthreads();
  if (w == 0) {
    float mx = -1e30f;
    for (int j = 0; j < WN; ++j) mx = fmaxf(mx, sc[j]);
    float sum = 0.f;
    for (int j = 0; j < WN; ++j) { sc[j] = __expf(sc[j] - mx); sum += sc[j]; }
    float inv = 1.f / sum;
    for (int j = 0; j < WN; ++j) sc[j] *= inv;
  }
  __syncthreads();
  if (w < WN) gates[w * NB + b] = (b < NC1) ? sc[w] : 0.f;
}

__global__ void k_qupd(float* __restrict__ q, const float* __restrict__ hf,
                       const float* __restrict__ hb) {
  int idx = blockIdx.x * blockDim.x + threadIdx.x;
  if (idx >= NB * HPAD) return;
  int b = idx / HPAD;
  if (b < NC1) q[idx] += hf[idx] + hb[idx];
}

// out[64][7]: row0 = utt[0], rows 1..63 = query[r-1]; dot with clf_W (7,300)
__global__ void k_clf(const float* __restrict__ utt, const float* __restrict__ q,
                      const float* __restrict__ W, const float* __restrict__ bias,
                      float* __restrict__ out) {
  int idx = blockIdx.x * blockDim.x + threadIdx.x;
  if (idx >= NB * 7) return;
  int c = idx % 7, r = idx / 7;
  const float* s = (r == 0) ? utt : (q + (size_t)(r - 1) * HPAD);
  float acc = bias[c];
  for (int k = 0; k < HDIM; ++k) acc += s[k] * W[c * HDIM + k];
  out[idx] = acc;
}

// ---------------------------------------------------------------------------
// Host orchestration
// ---------------------------------------------------------------------------
extern "C" void kernel_launch(void* const* d_in, const int* in_sizes, int n_in,
                              void* d_out, int out_size, void* d_ws, size_t ws_size,
                              hipStream_t stream) {
  const int*   ids   = (const int*)  d_in[0];
  const int*   slen  = (const int*)  d_in[1];
  const float* embed = (const float*)d_in[2];
  const float* Wih0  = (const float*)d_in[3];
  const float* Whh0  = (const float*)d_in[4];
  const float* bih0  = (const float*)d_in[5];
  const float* bhh0  = (const float*)d_in[6];
  const float* Wih1  = (const float*)d_in[7];
  const float* Whh1  = (const float*)d_in[8];
  const float* bih1  = (const float*)d_in[9];
  const float* bhh1  = (const float*)d_in[10];
  const float* linW  = (const float*)d_in[11];
  const float* linb  = (const float*)d_in[12];
  const float* cWih  = (const float*)d_in[13];
  const float* cWhh  = (const float*)d_in[14];
  const float* cbih  = (const float*)d_in[15];
  const float* cbhh  = (const float*)d_in[16];
  const float* attW  = (const float*)d_in[17];
  const float* attB  = (const float*)d_in[18];
  const float* clfW  = (const float*)d_in[19];
  const float* clfb  = (const float*)d_in[20];
  float* out = (float*)d_out;

  uint8_t* p = (uint8_t*)d_ws;
  auto take = [&](size_t bytes) -> uint8_t* {
    uint8_t* r = p;
    p += (bytes + 255) & ~(size_t)255;
    return r;
  };

  // Packed weights / biases
  bf16 *pWih0[2], *pWhh0[2], *pWih1[2], *pWhh1[2], *pcWih[2], *pcWhh[2];
  float *pbih0[2], *pbhh0[2], *pbih1[2], *pbhh1[2], *pcbih[2], *pcbhh[2];
  for (int d = 0; d < 2; ++d) {
    pWih0[d] = (bf16*)take((size_t)G3 * KPAD * 2);
    pWhh0[d] = (bf16*)take((size_t)G3 * KPAD * 2);
    pWih1[d] = (bf16*)take((size_t)G3 * K2PAD * 2);
    pWhh1[d] = (bf16*)take((size_t)G3 * KPAD * 2);
    pcWih[d] = (bf16*)take((size_t)G3 * KPAD * 2);
    pcWhh[d] = (bf16*)take((size_t)G3 * KPAD * 2);
    pbih0[d] = (float*)take(G3 * 4);
    pbhh0[d] = (float*)take(G3 * 4);
    pbih1[d] = (float*)take(G3 * 4);
    pbhh1[d] = (float*)take(G3 * 4);
    pcbih[d] = (float*)take(G3 * 4);
    pcbhh[d] = (float*)take(G3 * 4);
  }
  bf16*  pLinW = (bf16*) take((size_t)HPAD * K2PAD * 2);
  float* pLinB = (float*)take(HPAD * 4);
  bf16 *pAttC[3][2], *pAttU[3][2];
  float *pAttBc[3][2], *pAttBu[3][2];
  for (int h = 0; h < 3; ++h)
    for (int d = 0; d < 2; ++d) {
      pAttC[h][d]  = (bf16*) take((size_t)G2 * KPAD * 2);
      pAttU[h][d]  = (bf16*) take((size_t)G2 * KPAD * 2);
      pAttBc[h][d] = (float*)take(G2 * 4);
      pAttBu[h][d] = (float*)take(G2 * 4);
    }

  // Activations
  bf16*  X0   = (bf16*) take((size_t)TT * NB * KPAD * 2);
  bf16*  X1   = (bf16*) take((size_t)TT * NB * K2PAD * 2);
  float* giA  = (float*)take((size_t)TT * NB * G3 * 4);   // reused: gi l0/l1, gi ctx, ci att (fwd)
  float* giB  = (float*)take((size_t)TT * NB * G3 * 4);   // same (bwd)
  float* o2   = (float*)take((size_t)TT * NB * O2STR * 4);
  bf16*  uttM = (bf16*) take((size_t)NB * K2PAD * 2);
  float* uttP = (float*)take((size_t)NB * HPAD * 4);
  float* uttF = (float*)take((size_t)NB * HPAD * 4);
  bf16*  batc = (bf16*) take((size_t)WN * NB * KPAD * 2);
  float* fctx = (float*)take((size_t)WN * NB * HPAD * 4);
  float* bctx = (float*)take((size_t)WN * NB * HPAD * 4);
  float* msf  = (float*)take((size_t)WN * NB * HPAD * 4);
  bf16*  msb  = (bf16*) take((size_t)WN * NB * KPAD * 2);
  float* gat  = (float*)take((size_t)WN * NB * 4);
  float* qry  = (float*)take((size_t)NB * HPAD * 4);
  float* hF   = (float*)take((size_t)NB * HPAD * 4);
  float* hB   = (float*)take((size_t)NB * HPAD * 4);

  const int THR = 256;

  // ---- pack all weights ----
  for (int d = 0; d < 2; ++d) {
    k_pack_w<<<CEILDIV(G3 * KPAD, THR), THR, 0, stream>>>(Wih0 + (size_t)d * 900 * 300, pWih0[d], 300, HPAD, 300, KPAD, 3);
    k_pack_w<<<CEILDIV(G3 * KPAD, THR), THR, 0, stream>>>(Whh0 + (size_t)d * 900 * 300, pWhh0[d], 300, HPAD, 300, KPAD, 3);
    k_pack_w<<<CEILDIV(G3 * K2PAD, THR), THR, 0, stream>>>(Wih1 + (size_t)d * 900 * 600, pWih1[d], 300, HPAD, 600, K2PAD, 3);
    k_pack_w<<<CEILDIV(G3 * KPAD, THR), THR, 0, stream>>>(Whh1 + (size_t)d * 900 * 300, pWhh1[d], 300, HPAD, 300, KPAD, 3);
    k_pack_w<<<CEILDIV(G3 * KPAD, THR), THR, 0, stream>>>(cWih + (size_t)d * 900 * 300, pcWih[d], 300, HPAD, 300, KPAD, 3);
    k_pack_w<<<CEILDIV(G3 * KPAD, THR), THR, 0, stream>>>(cWhh + (size_t)d * 900 * 300, pcWhh[d], 300, HPAD, 300, KPAD, 3);
    k_pack_b<<<CEILDIV(G3, THR), THR, 0, stream>>>(bih0 + d * 900, pbih0[d], 300, HPAD, 3);
    k_pack_b<<<CEILDIV(G3, THR), THR, 0, stream>>>(bhh0 + d * 900, pbhh0[d], 300, HPAD, 3);
    k_pack_b<<<CEILDIV(G3, THR), THR, 0, stream>>>(bih1 + d * 900, pbih1[d], 300, HPAD, 3);
    k_pack_b<<<CEILDIV(G3, THR), THR, 0, stream>>>(bhh1 + d * 900, pbhh1[d], 300, HPAD, 3);
    k_pack_b<<<CEILDIV(G3, THR), THR, 0, stream>>>(cbih + d * 900, pcbih[d], 300, HPAD, 3);
    k_pack_b<<<CEILDIV(G3, THR), THR, 0, stream>>>(cbhh + d * 900, pcbhh[d], 300, HPAD, 3);
  }
  k_pack_w<<<CEILDIV(HPAD * K2PAD, THR), THR, 0, stream>>>(linW, pLinW, 300, HPAD, 600, K2PAD, 1);
  k_pack_b<<<CEILDIV(HPAD, THR), THR, 0, stream>>>(linb, pLinB, 300, HPAD, 1);
  for (int h = 0; h < 3; ++h)
    for (int d = 0; d < 2; ++d) {
      const float* wb = attW + (((size_t)h * 2 + d) * 4) * 300 * 300;
      const float* bb = attB + (((size_t)h * 2 + d) * 4) * 300;
      k_pack_w<<<CEILDIV(HPAD * KPAD, THR), THR, 0, stream>>>(wb + 0 * 90000, pAttC[h][d], 300, HPAD, 300, KPAD, 1);
      k_pack_w<<<CEILDIV(HPAD * KPAD, THR), THR, 0, stream>>>(wb + 2 * 90000, pAttC[h][d] + (size_t)HPAD * KPAD, 300, HPAD, 300, KPAD, 1);
      k_pack_w<<<CEILDIV(HPAD * KPAD, THR), THR, 0, stream>>>(wb + 1 * 90000, pAttU[h][d], 300, HPAD, 300, KPAD, 1);
      k_pack_w<<<CEILDIV(HPAD * KPAD, THR), THR, 0, stream>>>(wb + 3 * 90000, pAttU[h][d] + (size_t)HPAD * KPAD, 300, HPAD, 300, KPAD, 1);
      k_pack_b<<<CEILDIV(HPAD, THR), THR, 0, stream>>>(bb + 0 * 300, pAttBc[h][d], 300, HPAD, 1);
      k_pack_b<<<CEILDIV(HPAD, THR), THR, 0, stream>>>(bb + 2 * 300, pAttBc[h][d] + HPAD, 300, HPAD, 1);
      k_pack_b<<<CEILDIV(HPAD, THR), THR, 0, stream>>>(bb + 1 * 300, pAttBu[h][d], 300, HPAD, 1);
      k_pack_b<<<CEILDIV(HPAD, THR), THR, 0, stream>>>(bb + 3 * 300, pAttBu[h][d] + HPAD, 300, HPAD, 1);
    }

  // ---- embedding + layer-0 input GEMMs ----
  k_embed<<<CEILDIV(TT * NB * KPAD, THR), THR, 0, stream>>>(ids, embed, X0);
  wmma_gemm_bias<3><<<dim3(TT * NB / 16, NT_H), 32, 0, stream>>>(X0, KPAD, pWih0[0], KPAD, pbih0[0], giA, G3, 10);
  wmma_gemm_bias<3><<<dim3(TT * NB / 16, NT_H), 32, 0, stream>>>(X0, KPAD, pWih0[1], KPAD, pbih0[1], giB, G3, 10);

  // ---- layer-0 scan (writes bf16 X1: fwd cols 0..299, bwd cols 300..599) ----
  k_fill_zero_bf16<<<CEILDIV(TT * NB * K2PAD, THR), THR, 0, stream>>>(X1, TT * NB * K2PAD);
  {
    ScanArgs a = {};
    a.gi[0] = giA; a.gi[1] = giB;
    a.whh[0] = pWhh0[0]; a.whh[1] = pWhh0[1];
    a.bhh[0] = pbhh0[0]; a.bhh[1] = pbhh0[1];
    a.slens = slen;
    a.obf[0] = X1; a.obf[1] = X1; a.obf_stride = K2PAD; a.obf_coff[0] = 0; a.obf_coff[1] = HDIM;
    a.T = TT; a.use_mask = 1;
    gru_scan<<<2, 1024, 0, stream>>>(a);
  }

  // ---- layer-1 input GEMMs + scan (writes f32 o2) ----
  wmma_gemm_bias<3><<<dim3(TT * NB / 16, NT_H), 32, 0, stream>>>(X1, K2PAD, pWih1[0], K2PAD, pbih1[0], giA, G3, 20);
  wmma_gemm_bias<3><<<dim3(TT * NB / 16, NT_H), 32, 0, stream>>>(X1, K2PAD, pWih1[1], K2PAD, pbih1[1], giB, G3, 20);
  {
    ScanArgs a = {};
    a.gi[0] = giA; a.gi[1] = giB;
    a.whh[0] = pWhh1[0]; a.whh[1] = pWhh1[1];
    a.bhh[0] = pbhh1[0]; a.bhh[1] = pbhh1[1];
    a.slens = slen;
    a.of32[0] = o2; a.of32[1] = o2; a.of32_stride = O2STR; a.of32_coff[0] = 0; a.of32_coff[1] = HDIM;
    a.T = TT; a.use_mask = 1;
    gru_scan<<<2, 1024, 0, stream>>>(a);
  }

  // ---- utterance vector: maxpool + linear + tanh ----
  k_maxpool<<<CEILDIV(NB * K2PAD, THR), THR, 0, stream>>>(o2, uttM);
  wmma_gemm_bias<1><<<dim3(NB / 16, NT_H), 32, 0, stream>>>(uttM, K2PAD, pLinW, K2PAD, pLinB, uttP, HPAD, 20);
  k_tanh_utt<<<CEILDIV(NB * HPAD, THR), THR, 0, stream>>>(uttP, uttF);

  // ---- context windows + ctx bi-GRU ----
  k_batches<<<CEILDIV(WN * NB * KPAD, THR), THR, 0, stream>>>(uttF, batc);
  wmma_gemm_bias<3><<<dim3(WN * NB / 16, NT_H), 32, 0, stream>>>(batc, KPAD, pcWih[0], KPAD, pcbih[0], giA, G3, 10);
  wmma_gemm_bias<3><<<dim3(WN * NB / 16, NT_H), 32, 0, stream>>>(batc, KPAD, pcWih[1], KPAD, pcbih[1], giB, G3, 10);
  {
    ScanArgs a = {};
    a.gi[0] = giA; a.gi[1] = giB;
    a.whh[0] = pcWhh[0]; a.whh[1] = pcWhh[1];
    a.bhh[0] = pcbhh[0]; a.bhh[1] = pcbhh[1];
    a.slens = slen;      // unused (use_mask=0)
    a.of32[0] = fctx; a.of32[1] = bctx; a.of32_stride = HPAD; a.of32_coff[0] = 0; a.of32_coff[1] = 0;
    a.T = WN; a.use_mask = 0;
    gru_scan<<<2, 1024, 0, stream>>>(a);
  }
  k_msum<<<CEILDIV(WN * NB * KPAD, THR), THR, 0, stream>>>(uttF, fctx, bctx, msf, msb);
  k_qinit<<<CEILDIV(NB * HPAD, THR), THR, 0, stream>>>(uttF, qry);

  // ---- attention hops ----
  for (int h = 0; h < 3; ++h) {
    k_scores<<<NB, 64, 0, stream>>>(qry, msf, gat);
    wmma_gemm_bias<2><<<dim3(WN * NB / 16, NT_H), 32, 0, stream>>>(msb, KPAD, pAttC[h][0], KPAD, pAttBc[h][0], giA, G2, 10);
    wmma_gemm_bias<2><<<dim3(WN * NB / 16, NT_H), 32, 0, stream>>>(msb, KPAD, pAttC[h][1], KPAD, pAttBc[h][1], giB, G2, 10);
    AttnArgs aa = {};
    aa.ci[0] = giA; aa.ci[1] = giB;
    aa.u[0] = pAttU[h][0]; aa.u[1] = pAttU[h][1];
    aa.bu[0] = pAttBu[h][0]; aa.bu[1] = pAttBu[h][1];
    aa.gates = gat;
    aa.hout[0] = hF; aa.hout[1] = hB;
    aa.T = WN;
    attn_scan<<<2, 1024, 0, stream>>>(aa);
    k_qupd<<<CEILDIV(NB * HPAD, THR), THR, 0, stream>>>(qry, hF, hB);
  }

  // ---- classifier ----
  k_clf<<<CEILDIV(NB * 7, THR), THR, 0, stream>>>(uttF, qry, clfW, clfb, out);
}
